// AttentionHead_50663434224523
// MI455X (gfx1250) — matrix operand order repaired and checked by reading, not compiled, not run on hardware
//
#include <hip/hip_runtime.h>
#include <hip/hip_bf16.h>

typedef __attribute__((ext_vector_type(16))) __bf16 bf16x16;
typedef __attribute__((ext_vector_type(8)))  __bf16 bf16x8;
typedef __attribute__((ext_vector_type(8)))  float  f32x8;

#define F8_ZERO {0.f,0.f,0.f,0.f,0.f,0.f,0.f,0.f}
#define LOG_ROPE_BASE 9.210340371976184f   // ln(10000)
#define SM_SCALE 0.08838834764831845f      // 1/sqrt(128), folded into Q

union AFrag { bf16x16 v; bf16x8 h[2]; };

__device__ __forceinline__ f32x8 wmma_bf16(bf16x16 a, bf16x16 b, f32x8 c) {
  return __builtin_amdgcn_wmma_f32_16x16x32_bf16(false, a, false, b, (short)0, c, false, false);
}

// reductions across a 16-lane half (rows of a WMMA C tile live across 16 lanes)
__device__ __forceinline__ float half16_max(float v) {
  v = fmaxf(v, __shfl_xor(v, 1, 32));
  v = fmaxf(v, __shfl_xor(v, 2, 32));
  v = fmaxf(v, __shfl_xor(v, 4, 32));
  v = fmaxf(v, __shfl_xor(v, 8, 32));
  return v;
}
__device__ __forceinline__ float half16_sum(float v) {
  v += __shfl_xor(v, 1, 32);
  v += __shfl_xor(v, 2, 32);
  v += __shfl_xor(v, 4, 32);
  v += __shfl_xor(v, 8, 32);
  return v;
}

// A fragment 16x32 (MxK) bf16, A row-major [M,lda]
__device__ __forceinline__ bf16x16 load_a_frag(const __bf16* A, int lda, int m0, int k0,
                                               int ln, int hf) {
  AFrag f;
  const __bf16* row = A + (size_t)(m0 + ln) * lda + k0;
  f.h[0] = *(const bf16x8*)(row + hf * 16);
  f.h[1] = *(const bf16x8*)(row + hf * 16 + 8);
  return f.v;
}
// B fragment 32x16 (KxN) where B(k,n) = W[n,k], W row-major [N,ldw]
__device__ __forceinline__ bf16x16 load_b_frag(const __bf16* W, int ldw, int n0, int k0,
                                               int ln, int hf) {
  return *(const bf16x16*)(W + (size_t)(n0 + ln) * ldw + k0 + hf * 16);
}

__global__ __launch_bounds__(256) void cvt_f32_bf16(const float* __restrict__ in,
                                                    __bf16* __restrict__ out, int n) {
  int i = blockIdx.x * blockDim.x + threadIdx.x;
  int stride = gridDim.x * blockDim.x;
  for (; i < n; i += stride) out[i] = (__bf16)in[i];
}

// ---------------- QKV GEMM + fused RoPE epilogue ----------------
// x_bf16 [4096,2048] @ w_atten_bf16[6144,2048]^T -> q/k/v bf16 [B,H,T,D]
// Wave tile 32x64 (2 A-frags x 4 B-frags -> 8 wmma / 6 fragment loads per k-step).
// Q additionally scaled by 1/sqrt(D) so attention omits it.
__global__ __launch_bounds__(256)
void qkv_rope_kernel(const __bf16* __restrict__ xb, const __bf16* __restrict__ wab,
                     __bf16* __restrict__ qb, __bf16* __restrict__ kb,
                     __bf16* __restrict__ vb) {
  const int tid = threadIdx.x, w = tid >> 5, ln = tid & 15, hf = (tid >> 4) & 1;
  const int m0 = blockIdx.x * 64 + (w >> 2) * 32;
  const int n0 = blockIdx.y * 256 + (w & 3) * 64;

  f32x8 acc[2][4] = {{F8_ZERO, F8_ZERO, F8_ZERO, F8_ZERO},
                     {F8_ZERO, F8_ZERO, F8_ZERO, F8_ZERO}};
  for (int k0 = 0; k0 < 2048; k0 += 32) {
    bf16x16 a0 = load_a_frag(xb, 2048, m0, k0, ln, hf);
    bf16x16 a1 = load_a_frag(xb, 2048, m0 + 16, k0, ln, hf);
    #pragma unroll
    for (int j = 0; j < 4; ++j) {
      bf16x16 bj = load_b_frag(wab, 2048, n0 + j * 16, k0, ln, hf);
      acc[0][j] = wmma_bf16(a0, bj, acc[0][j]);
      acc[1][j] = wmma_bf16(a1, bj, acc[1][j]);
    }
  }

  #pragma unroll
  for (int i = 0; i < 2; ++i)
    #pragma unroll
    for (int j = 0; j < 4; ++j)
      #pragma unroll
      for (int r = 0; r < 8; ++r) {
        int gm = m0 + i * 16 + r + hf * 8;           // row in [B*T)
        int gn = n0 + j * 16 + ln;                   // col in [3C)
        float val = acc[i][j][r];
        float partner = __shfl_xor(val, 1, 32);      // RoPE pair = adjacent lane
        int bidx = gm >> 11, t = gm & 2047;
        int sect = gn >> 11, c = gn & 2047;
        int h = c >> 7, d = c & 127;
        if (sect < 2) {                              // rope on q and k (lane-uniform per tile)
          float freq = __expf(-(float)(d & ~1) * (LOG_ROPE_BASE / 128.0f));
          float sn, cs;
          __sincosf((float)t * freq, &sn, &cs);
          val = (d & 1) ? (val * cs + partner * sn) : (val * cs - partner * sn);
          if (sect == 0) val *= SM_SCALE;            // fold softmax scale into Q
        }
        __bf16* dst = (sect == 0) ? qb : ((sect == 1) ? kb : vb);
        dst[(((size_t)bidx * 16 + h) * 2048 + t) * 128 + d] = (__bf16)val;
      }
}

// ---------------- Flash attention (causal, online softmax) ----------------
__global__ __launch_bounds__(256)
void flash_attn_kernel(const __bf16* __restrict__ qb, const __bf16* __restrict__ kb,
                       const __bf16* __restrict__ vb, __bf16* __restrict__ yb) {
  __shared__ __align__(32) __bf16 sK[32 * 128];    // K tile, row-major [kv][d]
  __shared__ __align__(32) __bf16 sVt[128 * 32];   // V tile transposed [d][kv]
  __shared__ __align__(32) __bf16 sP[8][16 * 32];  // per-wave P transpose buffer

  const int tid = threadIdx.x, w = tid >> 5, ln = tid & 15, hf = (tid >> 4) & 1;
  const int qt = blockIdx.x & 15;
  const int bh = blockIdx.x >> 4;
  const int b = bh >> 4, h = bh & 15;
  const int q0 = qt * 128 + w * 16;                // this wave's 16 q rows
  const size_t head_off = ((size_t)b * 16 + h) * 2048 * 128;
  const __bf16* Q = qb + head_off;
  const __bf16* K = kb + head_off;
  const __bf16* V = vb + head_off;

  AFrag aq[4];
  {
    const __bf16* qrow = Q + (size_t)(q0 + ln) * 128;
    #pragma unroll
    for (int kk = 0; kk < 4; ++kk) {
      aq[kk].h[0] = *(const bf16x8*)(qrow + kk * 32 + hf * 16);
      aq[kk].h[1] = *(const bf16x8*)(qrow + kk * 32 + hf * 16 + 8);
    }
  }

  f32x8 o[8] = {F8_ZERO, F8_ZERO, F8_ZERO, F8_ZERO, F8_ZERO, F8_ZERO, F8_ZERO, F8_ZERO};
  float mrun[8], lrun[8];
  #pragma unroll
  for (int r = 0; r < 8; ++r) { mrun[r] = -1e30f; lrun[r] = 0.f; }

  const int lrow = tid >> 3;            // cooperative loads: 32 rows x 8 chunks of 16
  const int lcol = (tid & 7) * 16;
  const int kv_end = qt * 128 + 128;

  for (int kv0 = 0; kv0 < kv_end; kv0 += 32) {
    __syncthreads();  // previous tile's compute done before overwrite
    bf16x16 kt = *(const bf16x16*)(K + (size_t)(kv0 + lrow) * 128 + lcol);
    *(bf16x16*)(sK + lrow * 128 + lcol) = kt;
    bf16x16 vt = *(const bf16x16*)(V + (size_t)(kv0 + lrow) * 128 + lcol);
    #pragma unroll
    for (int e = 0; e < 16; ++e) sVt[(lcol + e) * 32 + lrow] = vt[e];
    __syncthreads();

    if (kv0 > q0 + 15) continue;        // wave-uniform: tile entirely above diagonal

    // S = Q K^T over D=128 (4 k-steps), two 16-col subtiles
    f32x8 s[2] = {F8_ZERO, F8_ZERO};
    #pragma unroll
    for (int kk = 0; kk < 4; ++kk) {
      #pragma unroll
      for (int j = 0; j < 2; ++j) {
        bf16x16 bk = *(const bf16x16*)(sK + (j * 16 + ln) * 128 + kk * 32 + hf * 16);
        s[j] = wmma_bf16(aq[kk].v, bk, s[j]);
      }
    }

    // online softmax per row (rows spread across 16-lane halves)
    const bool diag = (kv0 + 31 > q0);  // wave-uniform: tile touches the diagonal
    float alpha[8];
    __bf16* pw = &sP[w][0];
    #pragma unroll
    for (int r = 0; r < 8; ++r) {
      int trow = q0 + r + hf * 8;
      float s0 = s[0][r];
      float s1 = s[1][r];
      if (diag) {                        // causal mask only on diagonal tiles
        if (kv0 + ln > trow)      s0 = -1e30f;
        if (kv0 + 16 + ln > trow) s1 = -1e30f;
      }
      float mnew = fmaxf(mrun[r], half16_max(fmaxf(s0, s1)));
      float p0 = __expf(s0 - mnew);
      float p1 = __expf(s1 - mnew);
      float rs = half16_sum(p0 + p1);
      alpha[r] = __expf(mrun[r] - mnew);
      lrun[r] = lrun[r] * alpha[r] + rs;
      mrun[r] = mnew;
      pw[(r + hf * 8) * 32 + ln]      = (__bf16)p0;   // C-layout -> LDS
      pw[(r + hf * 8) * 32 + 16 + ln] = (__bf16)p1;
    }

    // re-read P in A-fragment layout (16x32)
    AFrag ap;
    ap.h[0] = *(const bf16x8*)(pw + ln * 32 + hf * 16);
    ap.h[1] = *(const bf16x8*)(pw + ln * 32 + hf * 16 + 8);

    // O = diag(alpha) O + P V   (8 d-subtiles of 16)
    #pragma unroll
    for (int dc = 0; dc < 8; ++dc) {
      #pragma unroll
      for (int r = 0; r < 8; ++r) o[dc][r] *= alpha[r];
      bf16x16 bv = *(const bf16x16*)(sVt + (dc * 16 + ln) * 32 + hf * 16);
      o[dc] = wmma_bf16(ap.v, bv, o[dc]);
    }
  }

  #pragma unroll
  for (int dc = 0; dc < 8; ++dc)
    #pragma unroll
    for (int r = 0; r < 8; ++r) {
      int trow = q0 + r + hf * 8;
      int d = dc * 16 + ln;
      float val = o[dc][r] / lrun[r];
      yb[((size_t)b * 2048 + trow) * 2048 + h * 128 + d] = (__bf16)val;
    }
}

// ---------------- Projection GEMM -> fp32 output ----------------
__global__ __launch_bounds__(256)
void proj_kernel(const __bf16* __restrict__ yb, const __bf16* __restrict__ wpb,
                 float* __restrict__ out) {
  const int tid = threadIdx.x, w = tid >> 5, ln = tid & 15, hf = (tid >> 4) & 1;
  const int m0 = blockIdx.x * 64 + (w >> 2) * 32;
  const int n0 = blockIdx.y * 256 + (w & 3) * 64;

  f32x8 acc[2][4] = {{F8_ZERO, F8_ZERO, F8_ZERO, F8_ZERO},
                     {F8_ZERO, F8_ZERO, F8_ZERO, F8_ZERO}};
  for (int k0 = 0; k0 < 2048; k0 += 32) {
    bf16x16 a0 = load_a_frag(yb, 2048, m0, k0, ln, hf);
    bf16x16 a1 = load_a_frag(yb, 2048, m0 + 16, k0, ln, hf);
    #pragma unroll
    for (int j = 0; j < 4; ++j) {
      bf16x16 bj = load_b_frag(wpb, 2048, n0 + j * 16, k0, ln, hf);
      acc[0][j] = wmma_bf16(a0, bj, acc[0][j]);
      acc[1][j] = wmma_bf16(a1, bj, acc[1][j]);
    }
  }
  #pragma unroll
  for (int i = 0; i < 2; ++i)
    #pragma unroll
    for (int j = 0; j < 4; ++j)
      #pragma unroll
      for (int r = 0; r < 8; ++r) {
        int gm = m0 + i * 16 + r + hf * 8;
        int gn = n0 + j * 16 + ln;
        out[(size_t)gm * 2048 + gn] = acc[i][j][r];
      }
}

extern "C" void kernel_launch(void* const* d_in, const int* in_sizes, int n_in,
                              void* d_out, int out_size, void* d_ws, size_t ws_size,
                              hipStream_t stream) {
  (void)in_sizes; (void)n_in; (void)out_size; (void)ws_size;
  const float* x  = (const float*)d_in[0];   // [2,2048,2048]
  const float* wa = (const float*)d_in[1];   // [6144,2048]
  const float* wp = (const float*)d_in[2];   // [2048,2048]

  char* ws = (char*)d_ws;
  const size_t SZ_X  = (size_t)4096 * 2048 * 2;   // 16 MiB
  const size_t SZ_WA = (size_t)6144 * 2048 * 2;   // 24 MiB
  const size_t SZ_WP = (size_t)2048 * 2048 * 2;   // 8 MiB
  const size_t SZ_H  = (size_t)2 * 16 * 2048 * 128 * 2;  // 16 MiB each

  __bf16* xb  = (__bf16*)(ws);                    // reused as yb after attention
  __bf16* wab = (__bf16*)(ws + SZ_X);
  __bf16* wpb = (__bf16*)(ws + SZ_X + SZ_WA);
  __bf16* qb  = (__bf16*)(ws + SZ_X + SZ_WA + SZ_WP);
  __bf16* kb  = (__bf16*)(ws + SZ_X + SZ_WA + SZ_WP + SZ_H);
  __bf16* vb  = (__bf16*)(ws + SZ_X + SZ_WA + SZ_WP + 2 * SZ_H);
  __bf16* yb  = xb;  // x no longer needed once QKV is produced

  cvt_f32_bf16<<<2048, 256, 0, stream>>>(x,  xb,  4096 * 2048);
  cvt_f32_bf16<<<2048, 256, 0, stream>>>(wa, wab, 6144 * 2048);
  cvt_f32_bf16<<<2048, 256, 0, stream>>>(wp, wpb, 2048 * 2048);

  qkv_rope_kernel<<<dim3(64, 24), 256, 0, stream>>>(xb, wab, qb, kb, vb);
  flash_attn_kernel<<<512, 256, 0, stream>>>(qb, kb, vb, yb);
  proj_kernel<<<dim3(64, 8), 256, 0, stream>>>(yb, wpb, (float*)d_out);
}
